// MultiHeadAttention_69269232550347
// MI455X (gfx1250) — compile-verified, matrix operand order checked by reading, not run on hardware
//
#include <hip/hip_runtime.h>
#include <hip/hip_bf16.h>

// ---------------------------------------------------------------------------
// MultiHeadAttention forward for MI455X (gfx1250, wave32, WMMA).
// B=2, S=2048, E=1024, H=16, D=64.
// Outputs (concatenated in d_out): output [B,S,E] f32, attn_weights [B,H,S,S] f32.
//
// Design (memory-roofline driven):
//  * attn_weights (537 MB f32) is a mandatory output -> problem is bound by a
//    single pass over the S x S matrix. attn_fused keeps a 32x2048 f32 score
//    block in 256 KB LDS (CDNA5: 320 KB/WGP) and writes attn exactly once.
//  * all GEMMs use v_wmma_f32_16x16x32_bf16 (f32 accum), wave32, 8-wave blocks.
//  * projection GEMMs stage the shared weight k-tile into LDS with
//    GLOBAL_LOAD_ASYNC_TO_LDS_B128 (ASYNCcnt), double buffered -> B tile read
//    from L2 once per block instead of once per M-wave (4x less weight traffic).
// ---------------------------------------------------------------------------

#define E_DIM  1024
#define HEADS  16
#define HDIM   64
#define BATCH  2
#define SEQ    2048
#define MTOT   (BATCH * SEQ)   // 4096 token rows

typedef __attribute__((ext_vector_type(16))) __bf16 v16bf;
typedef __attribute__((ext_vector_type(8)))  float  v8f;

static __device__ inline v8f wmma_bf16(v16bf a, v16bf b, v8f c) {
  // D(f32 16x16) = A(16x32 bf16) * B(32x16 bf16) + C
  return __builtin_amdgcn_wmma_f32_16x16x32_bf16(
      /*neg_a=*/false, a, /*neg_b=*/false, b,
      /*c_mod=*/(short)0, c, /*reuse_a=*/false, /*reuse_b=*/false);
}

static __device__ inline v16bf load16bf(const __bf16* p) {
  return *(const v16bf*)p;          // two global_load_b128 (or ds_load_b128)
}

static __device__ inline v16bf cvt16(const float* p) {
  v16bf r;
#pragma unroll
  for (int t = 0; t < 16; ++t) r[t] = (__bf16)p[t];
  return r;
}

// Async copy: 16 bytes global -> LDS per lane, tracked with ASYNCcnt.
// VGLOBAL async field order: VDST = per-lane LDS byte offset, ADDR = global VA.
static __device__ inline void async_ld16(uint32_t ldsOff, const void* g) {
  asm volatile("global_load_async_to_lds_b128 %0, %1, off"
               :: "v"(ldsOff), "v"(g)
               : "memory");
}
static __device__ inline void wait_async_le1() {
  asm volatile("s_wait_asynccnt 0x1" ::: "memory");
}
static __device__ inline void wait_async_le0() {
  asm volatile("s_wait_asynccnt 0x0" ::: "memory");
}

// ---------------------------------------------------------------------------
// f32 -> bf16 conversion (inputs + weights), one-time, bandwidth-trivial.
// ---------------------------------------------------------------------------
__global__ void cvt_f32_bf16(const float* __restrict__ in,
                             __bf16* __restrict__ out, int n) {
  int i = blockIdx.x * 256 + threadIdx.x;
  if (i < n) out[i] = (__bf16)in[i];
}

// ---------------------------------------------------------------------------
// Shared NT-GEMM core: acc[2][2] += X[m,k] * W[n,k], K = E_DIM, k-step 32.
// Block tile 128(M) x 64(N); 8 waves: wm = wid&3 (M), wn = wid>>2 (N).
// A streamed from global per wave; B (weights) staged via async-to-LDS,
// double buffered (2 x 4 KB), one async b128 per thread per k-tile.
// ---------------------------------------------------------------------------
static __device__ inline void gemm_core(const __bf16* __restrict__ X,
                                        const __bf16* __restrict__ W,
                                        int m0, int nblk,
                                        int tid, int wn, int lm, int lh,
                                        __bf16* bsh,     // [2][64*32]
                                        v8f acc[2][2]) {
  const int r   = tid >> 2;                 // 0..63: B-tile row (output col)
  const int seg = tid & 3;                  // 16-byte segment of the 32-k row
  const __bf16*  gsrc      = W + (size_t)(nblk + r) * E_DIM + seg * 8;
  const uint32_t ldsRowOff = (uint32_t)(r * 32 + seg * 8) * 2;   // bytes
  const uint32_t bufStride = 64 * 32 * 2;                        // 4096 B
  // Generic pointer to __shared__: low 32 bits are the LDS byte offset.
  const uint32_t bshBase   = (uint32_t)(uintptr_t)bsh;

  const __bf16* ap = X + (size_t)(m0 + lm) * E_DIM + lh * 16;

  // Prologue: stage k-tile 0 into buffer 0.
  async_ld16(bshBase + ldsRowOff, gsrc);

  const int NK = E_DIM / 32;
  for (int it = 0; it < NK; ++it) {
    const int kk = it * 32;
    if (it + 1 < NK) {                       // issue next tile, wait for current
      async_ld16(bshBase + (uint32_t)((it + 1) & 1) * bufStride + ldsRowOff,
                 gsrc + kk + 32);
      wait_async_le1();
    } else {
      wait_async_le0();
    }
    __syncthreads();                         // all waves' copies visible

    __builtin_prefetch(ap + kk + 64, 0, 3);  // near-scope prefetch of A stream
    v16bf a0 = load16bf(ap + kk);
    v16bf a1 = load16bf(ap + (size_t)16 * E_DIM + kk);
    const __bf16* bb = bsh + (size_t)(it & 1) * (64 * 32);
    v16bf b0 = load16bf(bb + (wn * 32 + lm) * 32 + lh * 16);        // ds_load
    v16bf b1 = load16bf(bb + (wn * 32 + 16 + lm) * 32 + lh * 16);   // ds_load
    acc[0][0] = wmma_bf16(a0, b0, acc[0][0]);
    acc[0][1] = wmma_bf16(a0, b1, acc[0][1]);
    acc[1][0] = wmma_bf16(a1, b0, acc[1][0]);
    acc[1][1] = wmma_bf16(a1, b1, acc[1][1]);

    __syncthreads();                         // buffer reusable next iteration
  }
}

// ---------------------------------------------------------------------------
// QKV projection: Y[m,n] = sum_k X[m,k] * W[n,k] + bias[n]   (NT GEMM)
// blockIdx.z selects Q / K / V. Q,K stored [B,H,S,D] bf16; V stored
// transposed [B,H,D,S] bf16 so the later P·V GEMM is NT (K-contiguous).
// ---------------------------------------------------------------------------
__global__ void qkv_gemm(const __bf16* __restrict__ xq, const __bf16* __restrict__ xk,
                         const __bf16* __restrict__ xv,
                         const __bf16* __restrict__ wq, const __bf16* __restrict__ wk,
                         const __bf16* __restrict__ wv,
                         const float* __restrict__ bq, const float* __restrict__ bk,
                         const float* __restrict__ bv,
                         __bf16* __restrict__ Qo, __bf16* __restrict__ Ko,
                         __bf16* __restrict__ Vo) {
  __shared__ __bf16 bsh[2][64 * 32];
  const int tid  = threadIdx.x;
  const int lane = tid & 31, wid = tid >> 5;
  const int wm = wid & 3, wn = wid >> 2;      // 4x2 wave grid
  const int lm = lane & 15, lh = lane >> 4;   // fragment lane coords

  const __bf16* X; const __bf16* W; const float* bias; __bf16* out; int vmode;
  if (blockIdx.z == 0)      { X = xq; W = wq; bias = bq; out = Qo; vmode = 0; }
  else if (blockIdx.z == 1) { X = xk; W = wk; bias = bk; out = Ko; vmode = 0; }
  else                      { X = xv; W = wv; bias = bv; out = Vo; vmode = 1; }

  const int m0   = blockIdx.x * 128 + wm * 32;
  const int nblk = blockIdx.y * 64;
  const int n0   = nblk + wn * 32;

  v8f acc[2][2] = {};
  gemm_core(X, W, m0, nblk, tid, wn, lm, lh, &bsh[0][0], acc);

#pragma unroll
  for (int mi = 0; mi < 2; ++mi)
#pragma unroll
    for (int ni = 0; ni < 2; ++ni)
#pragma unroll
      for (int i = 0; i < 8; ++i) {
        // C/D layout: VGPR i -> M=i (lanes 0-15) or M=i+8 (lanes 16-31), N=lm
        int row = m0 + mi * 16 + (lh ? 8 : 0) + i;
        int col = n0 + ni * 16 + lm;
        float v = acc[mi][ni][i] + bias[col];
        int b = row >> 11, s = row & (SEQ - 1);    // S = 2048
        int h = col >> 6,  d = col & (HDIM - 1);   // D = 64
        size_t idx = vmode
            ? ((size_t)((b * HEADS + h) * HDIM + d) * SEQ + s)   // V: [B,H,D,S]
            : ((size_t)((b * HEADS + h) * SEQ + s) * HDIM + d);  // Q,K: [B,H,S,D]
        out[idx] = (__bf16)v;
      }
}

// ---------------------------------------------------------------------------
// Fused scores -> softmax -> P·V.
// One workgroup per (b,h, 32-row block). The 32x2048 f32 score block lives in
// 256 KB LDS. attn_weights written to HBM exactly once.
// grid = (SEQ/32, B*H), block = 256 threads (8 waves), dynLDS = 32*SEQ*4 B.
// ---------------------------------------------------------------------------
__global__ void attn_fused(const __bf16* __restrict__ Q, const __bf16* __restrict__ K,
                           const __bf16* __restrict__ Vt,
                           float* __restrict__ attn, __bf16* __restrict__ ao) {
  extern __shared__ float sc[];               // [32][SEQ]
  const int tid  = threadIdx.x;
  const int lane = tid & 31, wid = tid >> 5;
  const int lm = lane & 15, lh = lane >> 4;
  const int bh = blockIdx.y;                  // b*H + h
  const int i0 = blockIdx.x * 32;             // first query row of this block
  const int b  = bh >> 4, h = bh & (HEADS - 1);

  const __bf16* Qb = Q  + (size_t)bh * SEQ * HDIM;   // [S,D]
  const __bf16* Kb = K  + (size_t)bh * SEQ * HDIM;   // [S,D]
  const __bf16* Vb = Vt + (size_t)bh * HDIM * SEQ;   // [D,S]

  // ---- Phase 1: scores (Q·Kᵀ)/8 into LDS --------------------------------
  {
    const int wm = wid & 1;                   // which 16-row tile of the 32
    const int j0 = (wid >> 1) * 512;          // 512-col slab per wave (32 n-tiles)
    const __bf16* aq = Qb + (size_t)(i0 + wm * 16 + lm) * HDIM + lh * 16;
    v16bf a0 = load16bf(aq);                  // k = 0..31
    v16bf a1 = load16bf(aq + 32);             // k = 32..63
    for (int nt = 0; nt < 32; ++nt) {
      int jc = j0 + nt * 16;
      const __bf16* bkp = Kb + (size_t)(jc + lm) * HDIM + lh * 16;
      __builtin_prefetch(bkp + 16 * HDIM, 0, 3);
      v16bf b0 = load16bf(bkp);
      v16bf b1 = load16bf(bkp + 32);
      v8f s = {};
      s = wmma_bf16(a0, b0, s);
      s = wmma_bf16(a1, b1, s);
#pragma unroll
      for (int i = 0; i < 8; ++i) {
        int r = wm * 16 + (lh ? 8 : 0) + i;
        sc[r * SEQ + jc + lm] = s[i] * 0.125f;   // 1/sqrt(64)
      }
    }
  }
  __syncthreads();

  // ---- Phase 2: row softmax in LDS, single write of attn_weights --------
  for (int r = wid; r < 32; r += 8) {         // one wave per row
    float* rp = sc + r * SEQ;
    float m = -3.0e38f;
    for (int j = lane; j < SEQ; j += 32) m = fmaxf(m, rp[j]);
#pragma unroll
    for (int off = 16; off > 0; off >>= 1) m = fmaxf(m, __shfl_xor(m, off, 32));
    float sum = 0.f;
    for (int j = lane; j < SEQ; j += 32) {
      float e = __expf(rp[j] - m);
      rp[j] = e;
      sum += e;
    }
#pragma unroll
    for (int off = 16; off > 0; off >>= 1) sum += __shfl_xor(sum, off, 32);
    float inv = 1.0f / sum;
    float* gp = attn + ((size_t)bh * SEQ + (size_t)(i0 + r)) * SEQ;
    for (int j = lane; j < SEQ; j += 32) {
      float w = rp[j] * inv;
      rp[j] = w;                               // keep normalized P in LDS
      gp[j] = w;                               // attn_weights output (f32)
    }
  }
  __syncthreads();

  // ---- Phase 3: attn_out = P(32x2048, LDS) @ V(2048x64) -----------------
  {
    const int wm = wid & 1;                    // 16-row tile
    const int wn = wid >> 1;                   // 0..3 -> 16-col tile of D=64
    v8f acc = {};
    const float*  ar = sc + (size_t)(wm * 16 + lm) * SEQ + lh * 16;
    const __bf16* br = Vb + (size_t)(wn * 16 + lm) * SEQ + lh * 16;
    for (int kk = 0; kk < SEQ; kk += 32) {
      __builtin_prefetch(br + kk + 64, 0, 3);
      v16bf a = cvt16(ar + kk);                // ds_load_b128 x4 + cvt to bf16
      v16bf v = load16bf(br + kk);
      acc = wmma_bf16(a, v, acc);
    }
#pragma unroll
    for (int i = 0; i < 8; ++i) {
      int s = i0 + wm * 16 + (lh ? 8 : 0) + i;
      int d = wn * 16 + lm;
      ao[((size_t)(b * SEQ + s)) * E_DIM + h * HDIM + d] = (__bf16)acc[i];
    }
  }
}

// ---------------------------------------------------------------------------
// Output projection: out[m,n] = sum_k ao[m,k] * Wo[n,k] + bo[n]  (f32 out)
// ---------------------------------------------------------------------------
__global__ void out_proj(const __bf16* __restrict__ A, const __bf16* __restrict__ W,
                         const float* __restrict__ bias, float* __restrict__ out) {
  __shared__ __bf16 bsh[2][64 * 32];
  const int tid  = threadIdx.x;
  const int lane = tid & 31, wid = tid >> 5;
  const int wm = wid & 3, wn = wid >> 2;
  const int lm = lane & 15, lh = lane >> 4;

  const int m0   = blockIdx.x * 128 + wm * 32;
  const int nblk = blockIdx.y * 64;
  const int n0   = nblk + wn * 32;

  v8f acc[2][2] = {};
  gemm_core(A, W, m0, nblk, tid, wn, lm, lh, &bsh[0][0], acc);

#pragma unroll
  for (int mi = 0; mi < 2; ++mi)
#pragma unroll
    for (int ni = 0; ni < 2; ++ni)
#pragma unroll
      for (int i = 0; i < 8; ++i) {
        int row = m0 + mi * 16 + (lh ? 8 : 0) + i;
        int col = n0 + ni * 16 + lm;
        out[(size_t)row * E_DIM + col] = acc[mi][ni][i] + bias[col];
      }
}

// ---------------------------------------------------------------------------
extern "C" void kernel_launch(void* const* d_in, const int* in_sizes, int n_in,
                              void* d_out, int out_size, void* d_ws, size_t ws_size,
                              hipStream_t stream) {
  (void)in_sizes; (void)n_in; (void)out_size; (void)ws_size;

  const float* query = (const float*)d_in[0];
  const float* key_  = (const float*)d_in[1];
  const float* value = (const float*)d_in[2];
  const float* Wq    = (const float*)d_in[3];
  const float* bq    = (const float*)d_in[4];
  const float* Wk    = (const float*)d_in[5];
  const float* bk    = (const float*)d_in[6];
  const float* Wv    = (const float*)d_in[7];
  const float* bv    = (const float*)d_in[8];
  const float* Wo    = (const float*)d_in[9];
  const float* bo    = (const float*)d_in[10];

  float* out  = (float*)d_out;                               // [B,S,E]
  float* attn = (float*)d_out + (size_t)BATCH * SEQ * E_DIM; // [B,H,S,S]

  // Workspace carve-up (bf16 buffers), ~64 MB total.
  char* ws = (char*)d_ws;
  size_t off = 0;
  auto carve = [&](size_t bytes) { char* p = ws + off; off += bytes; return p; };
  const size_t XE = (size_t)MTOT * E_DIM;    // 4M elems per activation tensor
  const size_t WE = (size_t)E_DIM * E_DIM;   // 1M elems per weight matrix
  __bf16* xq  = (__bf16*)carve(XE * 2);
  __bf16* xk  = (__bf16*)carve(XE * 2);
  __bf16* xv  = (__bf16*)carve(XE * 2);
  __bf16* wqb = (__bf16*)carve(WE * 2);
  __bf16* wkb = (__bf16*)carve(WE * 2);
  __bf16* wvb = (__bf16*)carve(WE * 2);
  __bf16* wob = (__bf16*)carve(WE * 2);
  __bf16* Qb  = (__bf16*)carve(XE * 2);      // [B,H,S,D]
  __bf16* Kb  = (__bf16*)carve(XE * 2);      // [B,H,S,D]
  __bf16* Vt  = (__bf16*)carve(XE * 2);      // [B,H,D,S] (pre-transposed)
  __bf16* ao  = (__bf16*)carve(XE * 2);      // [B,S,E]

  const int nX = (int)XE, nW = (int)WE;
  cvt_f32_bf16<<<(nX + 255) / 256, 256, 0, stream>>>(query, xq, nX);
  cvt_f32_bf16<<<(nX + 255) / 256, 256, 0, stream>>>(key_,  xk, nX);
  cvt_f32_bf16<<<(nX + 255) / 256, 256, 0, stream>>>(value, xv, nX);
  cvt_f32_bf16<<<(nW + 255) / 256, 256, 0, stream>>>(Wq, wqb, nW);
  cvt_f32_bf16<<<(nW + 255) / 256, 256, 0, stream>>>(Wk, wkb, nW);
  cvt_f32_bf16<<<(nW + 255) / 256, 256, 0, stream>>>(Wv, wvb, nW);
  cvt_f32_bf16<<<(nW + 255) / 256, 256, 0, stream>>>(Wo, wob, nW);

  qkv_gemm<<<dim3(MTOT / 128, E_DIM / 64, 3), 256, 0, stream>>>(
      xq, xk, xv, wqb, wkb, wvb, bq, bk, bv, Qb, Kb, Vt);

  const int ldsBytes = 32 * SEQ * (int)sizeof(float);   // 256 KB / WGP
  hipFuncSetAttribute(reinterpret_cast<const void*>(attn_fused),
                      hipFuncAttributeMaxDynamicSharedMemorySize, ldsBytes);
  attn_fused<<<dim3(SEQ / 32, BATCH * HEADS), 256, ldsBytes, stream>>>(
      Qb, Kb, Vt, attn, ao);

  out_proj<<<dim3(MTOT / 128, E_DIM / 64), 256, 0, stream>>>(ao, wob, bo, out);
}
